// SparseAttention_57939108823581
// MI455X (gfx1250) — matrix operand order repaired and checked by reading, not compiled, not run on hardware
//
#include <hip/hip_runtime.h>

typedef __bf16 v16bf __attribute__((ext_vector_type(16)));
typedef __bf16 v8bf  __attribute__((ext_vector_type(8)));
typedef float  v8f   __attribute__((ext_vector_type(8)));

// ---------------------------------------------------------------- helpers ---

__device__ __forceinline__ unsigned short f2bf(float f) {
  unsigned int u = __float_as_uint(f);
  unsigned int r = u + 0x7FFFu + ((u >> 16) & 1u);   // round-to-nearest-even
  return (unsigned short)(r >> 16);
}
__device__ __forceinline__ float bf2f(unsigned short h) {
  return __uint_as_float(((unsigned int)h) << 16);
}

// A-fragment (16x32 bf16): per-lane two 16B chunks at K-offset hh*8 and 16+hh*8
__device__ __forceinline__ v16bf ld_a(const unsigned short* p) {
  v8bf lo = *(const v8bf*)p;
  v8bf hi = *(const v8bf*)(p + 16);
  return __builtin_shufflevector(lo, hi, 0,1,2,3,4,5,6,7,8,9,10,11,12,13,14,15);
}
// B-fragment (32x16 bf16): per-lane two 16B chunks at K-offset hh*8 and 16+hh*8
__device__ __forceinline__ v16bf ld_b(const unsigned short* p) {
  v8bf lo = *(const v8bf*)p;
  v8bf hi = *(const v8bf*)(p + 16);
  return __builtin_shufflevector(lo, hi, 0,1,2,3,4,5,6,7,8,9,10,11,12,13,14,15);
}
__device__ __forceinline__ v8f wmma_bf16(v16bf a, v16bf b, v8f c) {
  return __builtin_amdgcn_wmma_f32_16x16x32_bf16(false, a, false, b, (short)0, c,
                                                 false, false);
}

// ------------------------------------------------ JAX Threefry-2x32 (key=42) -
__device__ __forceinline__ unsigned int rotl32(unsigned int x, int n) {
  return (x << n) | (x >> (32 - n));
}
__device__ __forceinline__ float tf_uniform(unsigned int i) {
  const unsigned int H = 16777216u;          // total 8*2048*2048 / 2
  bool first = i < H;
  unsigned int x0 = first ? i : (i - H);
  unsigned int x1 = first ? (i + H) : i;
  const unsigned int ks0 = 0u, ks1 = 42u, ks2 = 0x1BD11BDAu ^ 0u ^ 42u;
  x0 += ks0; x1 += ks1;
#define TFR(a,b,c,d_) { x0+=x1; x1=rotl32(x1,a); x1^=x0; \
                        x0+=x1; x1=rotl32(x1,b); x1^=x0; \
                        x0+=x1; x1=rotl32(x1,c); x1^=x0; \
                        x0+=x1; x1=rotl32(x1,d_); x1^=x0; }
  TFR(13,15,26,6);  x0 += ks1; x1 += ks2 + 1u;
  TFR(17,29,16,24); x0 += ks2; x1 += ks0 + 2u;
  TFR(13,15,26,6);  x0 += ks0; x1 += ks1 + 3u;
  TFR(17,29,16,24); x0 += ks1; x1 += ks2 + 4u;
  TFR(13,15,26,6);  x0 += ks2; x1 += ks0 + 5u;
#undef TFR
  unsigned int bits = first ? x0 : x1;
  return __uint_as_float((bits >> 9) | 0x3f800000u) - 1.0f;
}

// =================================================== kernel 1: QKV GEMMs ====
// y[m,e] = sum_d x[m,d] * W[e,d] + bias[e]; bf16 output. blockIdx.z: 0=Q,1=K,2=V
// Double-buffered LDS tiles: global loads for step ds+1 are issued before the
// WMMA block of step ds; conversion/stores happen after it; 1 barrier/step.
__global__ __launch_bounds__(256) void qkv_gemm_kernel(
    const float* __restrict__ x,
    const float* __restrict__ Wq, const float* __restrict__ bq,
    const float* __restrict__ Wk, const float* __restrict__ bk,
    const float* __restrict__ Wv, const float* __restrict__ bv,
    unsigned short* __restrict__ Qb, unsigned short* __restrict__ Kb,
    unsigned short* __restrict__ Vb)
{
  __shared__ unsigned short sA[2][128 * 40];   // x tile 128x32 bf16, stride 40
  __shared__ unsigned short sB[2][128 * 40];   // W tile 128x32 bf16, stride 40

  const int z = blockIdx.z;
  const float* W    = (z == 0) ? Wq : ((z == 1) ? Wk : Wv);
  const float* bias = (z == 0) ? bq : ((z == 1) ? bk : bv);
  unsigned short* O = (z == 0) ? Qb : ((z == 1) ? Kb : Vb);

  const int m0 = blockIdx.x * 128;
  const int e0 = blockIdx.y * 128;
  const int t  = threadIdx.x;
  const int l      = t & 31;
  const int wv_    = t >> 5;
  const int lane16 = l & 15;
  const int hh     = l >> 4;
  const int wm = wv_ & 3;      // 4 waves over M (32 rows each)
  const int wn = wv_ >> 2;     // 2 waves over N (64 cols each)

  // per-thread staging coordinates (4 float4 per matrix per step)
  int srow[4], sc4[4];
  const float *xp[4], *wp[4];
  for (int i = 0; i < 4; ++i) {
    int idx = t + i * 256;
    srow[i] = idx >> 3; sc4[i] = idx & 7;
    xp[i] = x + (size_t)(m0 + srow[i]) * 1024 + sc4[i] * 4;
    wp[i] = W + (size_t)(e0 + srow[i]) * 1024 + sc4[i] * 4;
  }

  float4 fx[4], fw[4];
  // prologue: stage tile 0 into buffer 0
  for (int i = 0; i < 4; ++i) { fx[i] = *(const float4*)xp[i];
                                fw[i] = *(const float4*)wp[i]; }
  for (int i = 0; i < 4; ++i) {
    unsigned int* pa = (unsigned int*)&sA[0][srow[i] * 40 + sc4[i] * 4];
    pa[0] = (unsigned)f2bf(fx[i].x) | ((unsigned)f2bf(fx[i].y) << 16);
    pa[1] = (unsigned)f2bf(fx[i].z) | ((unsigned)f2bf(fx[i].w) << 16);
    unsigned int* pb = (unsigned int*)&sB[0][srow[i] * 40 + sc4[i] * 4];
    pb[0] = (unsigned)f2bf(fw[i].x) | ((unsigned)f2bf(fw[i].y) << 16);
    pb[1] = (unsigned)f2bf(fw[i].z) | ((unsigned)f2bf(fw[i].w) << 16);
  }
  __syncthreads();

  v8f acc[2][4] = {};
  for (int ds = 0; ds < 32; ++ds) {
    const int cur = ds & 1;
    const bool hasNext = (ds + 1) < 32;
    if (hasNext) {                       // issue next-tile loads early
      for (int i = 0; i < 4; ++i) {
        fx[i] = *(const float4*)(xp[i] + (ds + 1) * 32);
        fw[i] = *(const float4*)(wp[i] + (ds + 1) * 32);
      }
    }

    v16bf af[2], bfr[4];
    for (int mi = 0; mi < 2; ++mi)
      af[mi] = ld_a(&sA[cur][(wm * 32 + mi * 16 + lane16) * 40 + hh * 8]);
    for (int ni = 0; ni < 4; ++ni)
      bfr[ni] = ld_b(&sB[cur][(wn * 64 + ni * 16 + lane16) * 40 + hh * 8]);
    for (int mi = 0; mi < 2; ++mi)
      for (int ni = 0; ni < 4; ++ni)
        acc[mi][ni] = wmma_bf16(af[mi], bfr[ni], acc[mi][ni]);

    if (hasNext) {                       // convert+store after compute
      const int nxt = cur ^ 1;
      for (int i = 0; i < 4; ++i) {
        unsigned int* pa = (unsigned int*)&sA[nxt][srow[i] * 40 + sc4[i] * 4];
        pa[0] = (unsigned)f2bf(fx[i].x) | ((unsigned)f2bf(fx[i].y) << 16);
        pa[1] = (unsigned)f2bf(fx[i].z) | ((unsigned)f2bf(fx[i].w) << 16);
        unsigned int* pb = (unsigned int*)&sB[nxt][srow[i] * 40 + sc4[i] * 4];
        pb[0] = (unsigned)f2bf(fw[i].x) | ((unsigned)f2bf(fw[i].y) << 16);
        pb[1] = (unsigned)f2bf(fw[i].z) | ((unsigned)f2bf(fw[i].w) << 16);
      }
    }
    __syncthreads();
  }

  for (int ni = 0; ni < 4; ++ni) {
    int e = e0 + wn * 64 + ni * 16 + lane16;
    float bval = bias[e];
    for (int mi = 0; mi < 2; ++mi)
      for (int v = 0; v < 8; ++v) {
        int m = m0 + wm * 32 + mi * 16 + hh * 8 + v;
        O[(size_t)m * 1024 + e] = f2bf(acc[mi][ni][v] + bval);
      }
  }
}

// ================================== kernel 2: V -> Vt[b][d][s] transpose ====
__global__ __launch_bounds__(256) void vtrans_kernel(
    const unsigned short* __restrict__ Vb, unsigned short* __restrict__ Vt)
{
  __shared__ unsigned short tile[32][33];
  const int b  = blockIdx.z;
  const int s0 = blockIdx.x * 32, d0 = blockIdx.y * 32;
  const int tx = threadIdx.x, ty = threadIdx.y;      // 32 x 8
  for (int j = 0; j < 4; ++j) {
    int row = ty + j * 8;
    tile[row][tx] = Vb[((size_t)b * 2048 + s0 + row) * 1024 + d0 + tx];
  }
  __syncthreads();
  for (int j = 0; j < 4; ++j) {
    int row = ty + j * 8;
    Vt[((size_t)b * 1024 + d0 + row) * 2048 + s0 + tx] = tile[tx][row];
  }
}

// ======================= kernel 3: fused masked-softmax attention ===========
// One workgroup per (batch, 32-query tile). 8 waves.
__global__ __launch_bounds__(256) void attn_kernel(
    const unsigned short* __restrict__ Qb, const unsigned short* __restrict__ Kb,
    const unsigned short* __restrict__ Vt, float* __restrict__ out)
{
  __shared__ unsigned short sQ[32 * 1032];   // Q tile  32x1024 bf16 (padded)
  __shared__ unsigned short sP[32 * 2056];   // scores -> probs, 32x2048 bf16
  __shared__ float rowInv[32];

  const int bidx = blockIdx.x >> 6;          // 64 q-tiles per batch
  const int q0   = (blockIdx.x & 63) * 32;
  const int t = threadIdx.x;
  const int l = t & 31, w = t >> 5;
  const int lane16 = l & 15, hh = l >> 4;

  // ---- Phase 0: stage Q tile into LDS (u32 copies) ----
  {
    const unsigned int* src =
        (const unsigned int*)(Qb + ((size_t)bidx * 2048 + q0) * 1024);
    unsigned int* dst = (unsigned int*)sQ;
    for (int i = 0; i < 64; ++i) {
      int idx = t + i * 256;                 // 0..16383 u32 slots
      int row = idx >> 9, c2 = idx & 511;
      dst[row * 516 + c2] = src[row * 512 + c2];
    }
  }
  __syncthreads();

  // ---- Phase 1: S = Q K^T * scale, threefry mask, store bf16 ----
  // process key tiles in pairs: 4 WMMA per 2 A + 2 B fragment loads
  const float scale = 0.03125f;              // 1/sqrt(1024)
  const unsigned short* qrow0 = sQ + lane16 * 1032 + hh * 8;
  const unsigned short* qrow1 = sQ + (16 + lane16) * 1032 + hh * 8;
  for (int np = 0; np < 8; ++np) {
    v8f acc00 = {}, acc01 = {}, acc10 = {}, acc11 = {};
    const int kloc = w * 256 + np * 32;
    const unsigned short* krow0 =
        Kb + ((size_t)bidx * 2048 + kloc + lane16) * 1024 + hh * 8;
    const unsigned short* krow1 = krow0 + (size_t)16 * 1024;
    for (int ds = 0; ds < 32; ++ds) {
      if (ds < 24) {                       // stream-ahead prefetch (512 B)
        __builtin_prefetch(krow0 + (ds + 8) * 32, 0, 3);
        __builtin_prefetch(krow1 + (ds + 8) * 32, 0, 3);
      }
      v16bf a0 = ld_a(qrow0 + ds * 32);
      v16bf a1 = ld_a(qrow1 + ds * 32);
      v16bf b0 = ld_b(krow0 + ds * 32);
      v16bf b1 = ld_b(krow1 + ds * 32);
      acc00 = wmma_bf16(a0, b0, acc00);
      acc01 = wmma_bf16(a0, b1, acc01);
      acc10 = wmma_bf16(a1, b0, acc10);
      acc11 = wmma_bf16(a1, b1, acc11);
    }
    const int k0 = kloc + lane16, k1 = k0 + 16;
    for (int v = 0; v < 8; ++v) {
      int M0 = hh * 8 + v, M1 = 16 + M0;
      unsigned int r0 = (unsigned)(bidx * 2048 + q0 + M0) * 2048u;
      unsigned int r1 = (unsigned)(bidx * 2048 + q0 + M1) * 2048u;
      float s;
      s = (tf_uniform(r0 + (unsigned)k0) > 0.7f) ? acc00[v] * scale : 0.0f;
      sP[M0 * 2056 + k0] = f2bf(s);
      s = (tf_uniform(r0 + (unsigned)k1) > 0.7f) ? acc01[v] * scale : 0.0f;
      sP[M0 * 2056 + k1] = f2bf(s);
      s = (tf_uniform(r1 + (unsigned)k0) > 0.7f) ? acc10[v] * scale : 0.0f;
      sP[M1 * 2056 + k0] = f2bf(s);
      s = (tf_uniform(r1 + (unsigned)k1) > 0.7f) ? acc11[v] * scale : 0.0f;
      sP[M1 * 2056 + k1] = f2bf(s);
    }
  }
  __syncthreads();

  // ---- Phase 2: softmax (unnormalized exp in LDS, 1/sum per row) ----
  for (int rr = 0; rr < 4; ++rr) {
    const int r = w * 4 + rr;
    unsigned short* prow = sP + r * 2056;
    float m = -3.0e38f;
    for (int ti = 0; ti < 64; ++ti) m = fmaxf(m, bf2f(prow[l + ti * 32]));
    for (int off = 16; off > 0; off >>= 1) m = fmaxf(m, __shfl_xor(m, off, 32));
    float sum = 0.0f;
    for (int ti = 0; ti < 64; ++ti) {
      int idx = l + ti * 32;
      float p = __expf(bf2f(prow[idx]) - m);
      sum += p;
      prow[idx] = f2bf(p);
    }
    for (int off = 16; off > 0; off >>= 1) sum += __shfl_xor(sum, off, 32);
    if (l == 0) rowInv[r] = 1.0f / sum;
  }
  __syncthreads();

  // ---- Phase 3: O = P V, scale rows by 1/sum; column tiles in pairs ----
  const unsigned short* prow0 = sP + lane16 * 2056 + hh * 8;
  const unsigned short* prow1 = sP + (16 + lane16) * 2056 + hh * 8;
  for (int np = 0; np < 4; ++np) {
    v8f acc00 = {}, acc01 = {}, acc10 = {}, acc11 = {};
    const int d0 = w * 128 + np * 32 + lane16;
    const int d1 = d0 + 16;
    const unsigned short* vrow0 =
        Vt + ((size_t)bidx * 1024 + d0) * 2048 + hh * 8;
    const unsigned short* vrow1 = vrow0 + (size_t)16 * 2048;
    for (int kt = 0; kt < 64; ++kt) {
      if (kt < 56) {
        __builtin_prefetch(vrow0 + (kt + 8) * 32, 0, 3);
        __builtin_prefetch(vrow1 + (kt + 8) * 32, 0, 3);
      }
      v16bf a0 = ld_a(prow0 + kt * 32);
      v16bf a1 = ld_a(prow1 + kt * 32);
      v16bf b0 = ld_b(vrow0 + kt * 32);
      v16bf b1 = ld_b(vrow1 + kt * 32);
      acc00 = wmma_bf16(a0, b0, acc00);
      acc01 = wmma_bf16(a0, b1, acc01);
      acc10 = wmma_bf16(a1, b0, acc10);
      acc11 = wmma_bf16(a1, b1, acc11);
    }
    for (int v = 0; v < 8; ++v) {
      int M0 = hh * 8 + v, M1 = 16 + M0;
      size_t ro0 = ((size_t)(bidx * 2048 + q0 + M0)) * 1024;
      size_t ro1 = ((size_t)(bidx * 2048 + q0 + M1)) * 1024;
      out[ro0 + d0] = acc00[v] * rowInv[M0];
      out[ro0 + d1] = acc01[v] * rowInv[M0];
      out[ro1 + d0] = acc10[v] * rowInv[M1];
      out[ro1 + d1] = acc11[v] * rowInv[M1];
    }
  }
}

// ============================================================== launcher ====
extern "C" void kernel_launch(void* const* d_in, const int* in_sizes, int n_in,
                              void* d_out, int out_size, void* d_ws, size_t ws_size,
                              hipStream_t stream) {
  (void)in_sizes; (void)n_in; (void)out_size; (void)ws_size;
  const float* x  = (const float*)d_in[0];
  const float* Wq = (const float*)d_in[1];
  const float* bq = (const float*)d_in[2];
  const float* Wk = (const float*)d_in[3];
  const float* bk = (const float*)d_in[4];
  const float* Wv = (const float*)d_in[5];
  const float* bv = (const float*)d_in[6];

  unsigned short* Qb = (unsigned short*)d_ws;
  unsigned short* Kb = Qb + (size_t)16384 * 1024;
  unsigned short* Vb = Kb + (size_t)16384 * 1024;
  unsigned short* Vt = Vb + (size_t)16384 * 1024;

  qkv_gemm_kernel<<<dim3(128, 8, 3), 256, 0, stream>>>(x, Wq, bq, Wk, bk, Wv, bv,
                                                       Qb, Kb, Vb);
  vtrans_kernel<<<dim3(64, 32, 8), dim3(32, 8), 0, stream>>>(Vb, Vt);
  attn_kernel<<<dim3(512), 256, 0, stream>>>(Qb, Kb, Vt, (float*)d_out);
}
